// IConv_5317169512859
// MI455X (gfx1250) — matrix operand order repaired and checked by reading, not compiled, not run on hardware
//
#include <hip/hip_runtime.h>

// ---------------------------------------------------------------------------
// Fused IConv for MI455X (gfx1250, wave32):
//   stage1: kernel_k^T[d,o] = relu( mhc^T (d x l) @ W_k (l x o) )   [bf16 WMMA]
//   stage2: out[o,t]       += kernel_k (o x d) @ pep_win (d x t)    [bf16 WMMA]
// One workgroup per batch element, 8 waves, wave w owns o in [16w, 16w+16).
// ---------------------------------------------------------------------------

typedef __attribute__((ext_vector_type(16))) __bf16 v16bf;
typedef __attribute__((ext_vector_type(8)))  float  v8f;

union FragAB { v16bf v; unsigned u[8]; };
union FragC  { v8f   v; float    f[8]; };

__device__ __forceinline__ unsigned pack_bf16(float a, float b) {
  union { __bf16 h[2]; unsigned u; } p;
  p.h[0] = (__bf16)a;
  p.h[1] = (__bf16)b;
  return p.u;
}

__device__ __forceinline__ v8f wmma_bf16(v16bf a, v16bf b, v8f c) {
  // (neg_a, A, neg_b, B, c_mod, C, reuse_a, reuse_b)
  return __builtin_amdgcn_wmma_f32_16x16x32_bf16(false, a, false, b,
                                                 (short)0, c, false, false);
}

#define PEP_LEN 15
#define DDIM    64
#define OC      128
#define KS      9
#define ML      34
#define LPAD    64   // l padded to one 2-step bf16-WMMA reduction
#define TOUT    7

__global__ __launch_bounds__(256) void iconv_wmma_kernel(
    const float* __restrict__ pep,   // [bs][15][64]
    const float* __restrict__ mhc,   // [bs][34][64]
    const float* __restrict__ wgt,   // [128][9][34]
    const float* __restrict__ bias,  // [128]
    float* __restrict__ out)         // [bs][128][7]
{
  __shared__ __bf16 sMhcT[DDIM * LPAD];   // [d][l]  (l contiguous, zero-padded) 8KB
  __shared__ __bf16 sPep[16 * DDIM];      // [row][d]                            2KB
  __shared__ __bf16 sKer[8][16 * DDIM];   // per-wave [o_local][d]              16KB
  __shared__ float  sBias[OC];

  const int n    = blockIdx.x;
  const int tid  = threadIdx.x;
  const int wave = tid >> 5;
  const int lane = tid & 31;
  const int lo   = lane & 15;   // N-lane / C-column index
  const int hi   = lane >> 4;   // half-wave select

  const int obase = wave * 16 + lo;   // o column for stage-1 B fragment

  // Warm L2/L0 for this wave's weight slice (global_prefetch_b8).
  __builtin_prefetch(wgt + (size_t)obase * (KS * ML), 0, 1);

  // ---- cooperative staging: zero-pad + f32->bf16 convert into LDS ----------
  for (int i = tid; i < (DDIM * LPAD) / 2; i += 256)
    ((unsigned*)sMhcT)[i] = 0u;
  if (tid < OC) sBias[tid] = bias[tid];

  const float* mhcn = mhc + (size_t)n * ML * DDIM;
  for (int i = tid; i < ML * DDIM; i += 256) {
    int l = i >> 6, d = i & 63;
    sMhcT[d * LPAD + l] = (__bf16)mhcn[i];          // transpose: [d][l]
  }
  const float* pepn = pep + (size_t)n * PEP_LEN * DDIM;
  for (int i = tid; i < PEP_LEN * DDIM; i += 256)
    sPep[i] = (__bf16)pepn[i];                       // [row][d]
  __syncthreads();

  // ---- stage-1 A fragments (mhc^T): M=d, K=l; k-invariant -> cache in regs -
  FragAB A1[4][2];
#pragma unroll
  for (int dt = 0; dt < 4; ++dt)
#pragma unroll
    for (int s = 0; s < 2; ++s)
#pragma unroll
      for (int v = 0; v < 8; ++v) {
        int l = s * 32 + ((v >> 2) << 4) + (hi << 3) + ((v & 3) << 1);
        int d = dt * 16 + lo;
        A1[dt][s].u[v] = *(const unsigned*)(sMhcT + d * LPAD + l);
      }

  FragC acc;
#pragma unroll
  for (int v = 0; v < 8; ++v) acc.f[v] = 0.0f;

  const int tRow = (lo < TOUT) ? lo : (TOUT - 1);  // dead columns clamp in-range

  for (int k = 0; k < KS; ++k) {
    // -- stage-1 B: W[o,k,l] as (K=l, N=o); f32 pair -> packed bf16 ----------
    FragAB B1[2];
#pragma unroll
    for (int s = 0; s < 2; ++s)
#pragma unroll
      for (int v = 0; v < 8; ++v) {
        int l = s * 32 + ((v >> 2) << 4) + (hi << 3) + ((v & 3) << 1);
        unsigned pk = 0u;
        if (l < ML) {  // l even, so (l, l+1) is a full valid pair iff l <= 32
          const float* wp = wgt + ((size_t)obase * KS + k) * ML + l;
          pk = pack_bf16(wp[0], wp[1]);
        }
        B1[s].u[v] = pk;
      }

    // -- stage 1: C1[d,o] = relu(mhc^T @ W_k); write bf16 ker tile to LDS ----
#pragma unroll
    for (int dt = 0; dt < 4; ++dt) {
      FragC c;
#pragma unroll
      for (int v = 0; v < 8; ++v) c.f[v] = 0.0f;
      c.v = wmma_bf16(A1[dt][0].v, B1[0].v, c.v);
      c.v = wmma_bf16(A1[dt][1].v, B1[1].v, c.v);
#pragma unroll
      for (int v = 0; v < 8; v += 2) {       // M=d consecutive over v -> b32 pair
        float f0 = fmaxf(c.f[v], 0.0f);
        float f1 = fmaxf(c.f[v + 1], 0.0f);
        int d = dt * 16 + (hi << 3) + v;
        *(unsigned*)(&sKer[wave][lo * DDIM + d]) = pack_bf16(f0, f1);
      }
    }

    // -- stage 2: acc[o,t] += ker_k (o x d) @ pep[t+k] (d x t) ---------------
    // Per-wave scratch; in-wave DS ordering guarantees store->load order.
#pragma unroll
    for (int s2 = 0; s2 < 2; ++s2) {
      FragAB A2, B2;
#pragma unroll
      for (int v = 0; v < 8; ++v) {
        int dA = s2 * 32 + ((v >> 2) << 4) + (hi << 3) + ((v & 3) << 1);
        A2.u[v] = *(const unsigned*)(&sKer[wave][lo * DDIM + dA]);
        int dB = s2 * 32 + ((v >> 2) << 4) + (hi << 3) + ((v & 3) << 1);
        B2.u[v] = *(const unsigned*)(sPep + (tRow + k) * DDIM + dB);
      }
      acc.v = wmma_bf16(A2.v, B2.v, acc.v);
    }
  }

  // ---- epilogue: o = 16*wave + 8*hi + v, t = lo (< 7) ----------------------
  if (lo < TOUT) {
#pragma unroll
    for (int v = 0; v < 8; ++v) {
      int o = wave * 16 + (hi << 3) + v;
      out[((size_t)n * OC + o) * TOUT + lo] = acc.f[v] + sBias[o];
    }
  }
}

extern "C" void kernel_launch(void* const* d_in, const int* in_sizes, int n_in,
                              void* d_out, int out_size, void* d_ws, size_t ws_size,
                              hipStream_t stream) {
  const float* pep  = (const float*)d_in[0];
  const float* mhc  = (const float*)d_in[1];
  const float* wgt  = (const float*)d_in[2];
  const float* bias = (const float*)d_in[3];
  float* out = (float*)d_out;

  const int bs = in_sizes[0] / (PEP_LEN * DDIM);   // 2048
  iconv_wmma_kernel<<<dim3(bs), dim3(256), 0, stream>>>(pep, mhc, wgt, bias, out);
}
